// EmergencyDecoder_10316511445816
// MI455X (gfx1250) — matrix operand order repaired and checked by reading, not compiled, not run on hardware
//
#include <hip/hip_runtime.h>
#include <hip/hip_bf16.h>

// ---------------------------------------------------------------------------
// EmergencyDecoder forward for MI455X (gfx1250, wave32, WMMA + TDM).
//
// Pipeline:
//   1) chunk_mean  : ct (T,8,128) -> cs (B*nc, 128) bf16   [TDM HBM->LDS,
//                    double-buffered 50x128 tiles, s_wait_tensorcnt]
//   2) gemm(WMMA)  : s   = relu(cs @ w1^T + b1)   (8792x256)  bf16
//   3) gemm(WMMA)  : seq = s @ w2^T + b2          (8792x512)  f32
//   4) embed_ln    : h = LN(seq+pos+tok)          (8800x512)  f32 + bf16
//   5) per layer:  q/k/v/kg/vg/qg GEMMs (WMMA), banded local attention (VALU,
//                  LDS-staged), global-token attention, wo GEMM, add+LN,
//                  MLP GEMMs (GELU-erf fused), add+LN
//   6) final_head  : LN(cls row) -> (8x11, 8x5) concatenated into d_out
// ---------------------------------------------------------------------------

typedef __bf16 bf16;
typedef __attribute__((ext_vector_type(8)))  bf16  v8bf;
typedef __attribute__((ext_vector_type(16))) bf16  v16bf;
typedef __attribute__((ext_vector_type(8)))  float v8f;
typedef __attribute__((ext_vector_type(4)))  unsigned int u32x4;
typedef __attribute__((ext_vector_type(8)))  int i32x8;
typedef __attribute__((ext_vector_type(4)))  int i32x4;

#define KD   512      // model dim D
#define NHD  64       // head dim
#define SEQ  1100     // S after cls concat (1099 chunks + 1 cls, %50 == 0)
#define NC   1099     // number of chunks
#define NB   8        // batch
#define NCH  22       // S / W
#define WW   50       // window W

enum { ACT_NONE = 0, ACT_RELU = 1, ACT_GELU = 2 };

// ---------------- WMMA fragment loaders (ISA 7.12.2 layouts) ----------------
// A (16x32 bf16): lane g = lane/16, M = lane%16.
//   VGPR0..3 hold K = 8g .. 8g+7 ; VGPR4..7 hold K = 16+8g .. 23+8g
__device__ __forceinline__ v16bf ld_fragA(const bf16* __restrict__ rowk, int g) {
  v8bf lo = *(const v8bf*)(rowk + 8 * g);
  v8bf hi = *(const v8bf*)(rowk + 16 + 8 * g);
  return __builtin_shufflevector(lo, hi, 0, 1, 2, 3, 4, 5, 6, 7,
                                 8, 9, 10, 11, 12, 13, 14, 15);
}
// B (32x16 bf16): lane g = lane/16, N = lane%16.
//   VGPR0..3 hold K = 8g .. 8g+7 ; VGPR4..7 hold K = 16+8g .. 23+8g
__device__ __forceinline__ v16bf ld_fragB(const bf16* __restrict__ rowk, int g) {
  v8bf lo = *(const v8bf*)(rowk + 8 * g);
  v8bf hi = *(const v8bf*)(rowk + 16 + 8 * g);
  return __builtin_shufflevector(lo, hi, 0, 1, 2, 3, 4, 5, 6, 7,
                                 8, 9, 10, 11, 12, 13, 14, 15);
}

// ---------------- Generic WMMA GEMM:  C = act((A @ W^T + bias) * alpha) -----
// A: (M x K) bf16 row-major, lda;  W: (N x K) bf16 row-major, ldb.
// Writes f32 (Cf) and/or bf16 (Cb) outputs, ldc. Workgroup = 8 waves,
// tile 128x64; each wave = 32x32 via 2x2 v_wmma_f32_16x16x32_bf16.
__global__ __launch_bounds__(256)
void gemm_bf16_wmma(const bf16* __restrict__ A, int lda,
                    const bf16* __restrict__ W, int ldb,
                    const float* __restrict__ bias,
                    float* __restrict__ Cf, bf16* __restrict__ Cb, int ldc,
                    int M, int N, int K, float alpha, int act) {
  const int lane = threadIdx.x & 31;
  const int wave = threadIdx.x >> 5;          // 0..7
  const int wm = wave & 3, wn = wave >> 2;    // 4 (M) x 2 (N)
  const int m0 = blockIdx.x * 128 + wm * 32;
  const int n0 = blockIdx.y * 64 + wn * 32;
  const int g = lane >> 4;
  const int l16 = lane & 15;

  int ra0 = m0 + l16;       if (ra0 >= M) ra0 = M - 1;   // clamp (no EXEC holes)
  int ra1 = m0 + 16 + l16;  if (ra1 >= M) ra1 = M - 1;
  int rb0 = n0 + l16;       if (rb0 >= N) rb0 = N - 1;
  int rb1 = n0 + 16 + l16;  if (rb1 >= N) rb1 = N - 1;
  const bf16* pa0 = A + (long)ra0 * lda;
  const bf16* pa1 = A + (long)ra1 * lda;
  const bf16* pb0 = W + (long)rb0 * ldb;
  const bf16* pb1 = W + (long)rb1 * ldb;

  v8f acc00 = {}, acc01 = {}, acc10 = {}, acc11 = {};
  for (int kb = 0; kb < K; kb += 32) {
    v16bf a0 = ld_fragA(pa0 + kb, g);
    v16bf a1 = ld_fragA(pa1 + kb, g);
    v16bf b0 = ld_fragB(pb0 + kb, g);
    v16bf b1 = ld_fragB(pb1 + kb, g);
    acc00 = __builtin_amdgcn_wmma_f32_16x16x32_bf16(false, a0, false, b0,
                                                    (short)0, acc00, false, false);
    acc01 = __builtin_amdgcn_wmma_f32_16x16x32_bf16(false, a0, false, b1,
                                                    (short)0, acc01, false, false);
    acc10 = __builtin_amdgcn_wmma_f32_16x16x32_bf16(false, a1, false, b0,
                                                    (short)0, acc10, false, false);
    acc11 = __builtin_amdgcn_wmma_f32_16x16x32_bf16(false, a1, false, b1,
                                                    (short)0, acc11, false, false);
  }

  // D layout: lane (m/8)*16 + n, VGPR m%8  ->  m = m0+ia*16+g*8+v, n = n0+ib*16+l16
#define STORE_TILE(ACC, IA, IB)                                              \
  do {                                                                       \
    int n = n0 + (IB)*16 + l16;                                              \
    if (n < N) {                                                             \
      float bv = bias ? bias[n] : 0.0f;                                      \
      _Pragma("unroll") for (int vv = 0; vv < 8; ++vv) {                     \
        int m = m0 + (IA)*16 + g * 8 + vv;                                   \
        if (m < M) {                                                         \
          float x = ((ACC)[vv] + bv) * alpha;                                \
          if (act == ACT_RELU) x = fmaxf(x, 0.0f);                           \
          else if (act == ACT_GELU)                                          \
            x = 0.5f * x * (1.0f + erff(x * 0.70710678118654752f));          \
          if (Cf) Cf[(long)m * ldc + n] = x;                                 \
          if (Cb) Cb[(long)m * ldc + n] = (bf16)x;                           \
        }                                                                    \
      }                                                                      \
    }                                                                        \
  } while (0)

  STORE_TILE(acc00, 0, 0);
  STORE_TILE(acc01, 0, 1);
  STORE_TILE(acc10, 1, 0);
  STORE_TILE(acc11, 1, 1);
#undef STORE_TILE
}

// ---------------- chunk mean via Tensor Data Mover --------------------------
// cs[b*nc+c][h] = mean_t ct[c*100+t][b][h].  Each block owns one (b,c) chunk:
// two 50x128 fp32 tiles (row stride 1024 elems) are DMA'd HBM->LDS with
// TENSOR_LOAD_TO_LDS and double-buffered against the VALU reduction.
#if __has_builtin(__builtin_amdgcn_tensor_load_to_lds)
#define HAVE_TDM 1
// D# groups per CDNA5 ISA 8.3/8.4 (2D tensor; groups 2/3 + trailing group
// zero => dims 2..4 unused). This toolchain exposes the 6-arg builtin:
//   (u32x4 g0, i32x8 g1, i32x4 g2, i32x4 g3, i32x8 gx, i32 cpol)
__device__ __forceinline__ void tdm_load_2d(const void* gaddr, unsigned lds_off,
                                            unsigned tile_d0, unsigned tile_d1,
                                            unsigned stride0_elems) {
  unsigned long long ga = (unsigned long long)gaddr;
  u32x4 g0;
  g0[0] = 1u;                                          // count=1 (valid, user)
  g0[1] = lds_off;                                     // lds_addr (bytes)
  g0[2] = (unsigned)(ga & 0xFFFFFFFFu);                // global_addr[31:0]
  g0[3] = (unsigned)((ga >> 32) & 0x1FFFFFFu)          // global_addr[56:32]
          | (2u << 30);                                // type=2 ("image")
  i32x8 g1;
  g1[0] = (int)(2u << 16);                             // data_size=2 (4B), mask=0
  g1[1] = (int)((tile_d0 & 0xFFFFu) << 16);            // tensor_dim0[15:0]
  g1[2] = (int)(((tile_d0 >> 16) & 0xFFFFu)            // tensor_dim0[31:16]
          | ((tile_d1 & 0xFFFFu) << 16));              // tensor_dim1[15:0]
  g1[3] = (int)(((tile_d1 >> 16) & 0xFFFFu)            // tensor_dim1[31:16]
          | ((tile_d0 & 0xFFFFu) << 16));              // tile_dim0
  g1[4] = (int)(tile_d1 & 0xFFFFu);                    // tile_dim1; tile_dim2=0
  g1[5] = (int)stride0_elems;                          // tensor_dim0_stride[31:0]
  g1[6] = 0;                                           // stride0[47:32], stride1 lo
  g1[7] = 0;                                           // stride1 hi
  i32x4 gz4 = {0, 0, 0, 0};
  i32x8 gz8 = {0, 0, 0, 0, 0, 0, 0, 0};
  __builtin_amdgcn_tensor_load_to_lds(g0, g1, gz4, gz4, gz8, 0);
}
#else
#define HAVE_TDM 0
#endif

__global__ __launch_bounds__(128)
void chunk_mean(const float* __restrict__ ct, bf16* __restrict__ cs, int nc) {
  int bc = blockIdx.x;            // b*nc + c
  int b = bc / nc, c = bc % nc;
  int h = threadIdx.x;            // 0..127
#if defined(__gfx1250__) && HAVE_TDM
  __shared__ float tile[2][50 * 128];       // 2 x 25.6 KB, double buffered
  const float* base = ct + ((long)(c * 100) * NB + b) * 128;
  if (threadIdx.x == 0) {
    tdm_load_2d(base, (unsigned)(size_t)&tile[0][0], 128, 50, 1024);
    tdm_load_2d(base + (long)50 * 1024, (unsigned)(size_t)&tile[1][0],
                128, 50, 1024);
    __builtin_amdgcn_s_wait_tensorcnt(1);   // first tile landed
  }
  __syncthreads();
  float s = 0.0f;
  for (int i = 0; i < 50; ++i) s += tile[0][i * 128 + h];
  if (threadIdx.x == 0) __builtin_amdgcn_s_wait_tensorcnt(0);
  __syncthreads();
  for (int i = 0; i < 50; ++i) s += tile[1][i * 128 + h];
#else
  const float* p = ct + ((long)(c * 100) * NB + b) * 128 + h;
  const long stride = (long)NB * 128;
  float s = 0.0f;
  for (int i = 0; i < 100; ++i) {
    if (i < 92) __builtin_prefetch(p + (i + 8) * stride, 0, 1);
    s += p[i * stride];
  }
#endif
  cs[(long)bc * 128 + h] = (bf16)(s * 0.01f);
}

// ---------------- f32 -> bf16 conversion -----------------------------------
__global__ __launch_bounds__(256)
void f2bf(const float* __restrict__ s, bf16* __restrict__ d, long n) {
  long i = (long)blockIdx.x * 256 + threadIdx.x;
  if (i < n) d[i] = (bf16)s[i];
}

// ---------------- embed + LN: h = LN(seq + pos + tok) ----------------------
__global__ __launch_bounds__(512)
void embed_ln(const float* __restrict__ seq, const float* __restrict__ cls,
              const float* __restrict__ pos, const float* __restrict__ tok,
              const float* __restrict__ gma, const float* __restrict__ bta,
              float* __restrict__ hf, bf16* __restrict__ hb) {
  int r = blockIdx.x;                 // b*S + s
  int b = r / SEQ, s = r % SEQ;
  int d = threadIdx.x;
  float x = (s < NC) ? seq[((long)(b * NC + s)) * KD + d] : cls[d];
  x += pos[(long)s * KD + d] + tok[d];
  __shared__ float red[KD];
  red[d] = x;
  __syncthreads();
  for (int o = 256; o > 0; o >>= 1) { if (d < o) red[d] += red[d + o]; __syncthreads(); }
  float mean = red[0] * (1.0f / KD);
  __syncthreads();
  float c = x - mean;
  red[d] = c * c;
  __syncthreads();
  for (int o = 256; o > 0; o >>= 1) { if (d < o) red[d] += red[d + o]; __syncthreads(); }
  float var = red[0] * (1.0f / KD);
  float y = c * rsqrtf(var + 1e-5f) * gma[d] + bta[d];
  hf[(long)r * KD + d] = y;
  hb[(long)r * KD + d] = (bf16)y;
}

// ---------------- residual + LN: h = LN(h + a) -----------------------------
__global__ __launch_bounds__(512)
void add_ln(float* __restrict__ hf, const float* __restrict__ a,
            const float* __restrict__ gma, const float* __restrict__ bta,
            bf16* __restrict__ hb) {
  int r = blockIdx.x;
  int d = threadIdx.x;
  float x = hf[(long)r * KD + d] + a[(long)r * KD + d];
  __shared__ float red[KD];
  red[d] = x;
  __syncthreads();
  for (int o = 256; o > 0; o >>= 1) { if (d < o) red[d] += red[d + o]; __syncthreads(); }
  float mean = red[0] * (1.0f / KD);
  __syncthreads();
  float c = x - mean;
  red[d] = c * c;
  __syncthreads();
  for (int o = 256; o > 0; o >>= 1) { if (d < o) red[d] += red[d + o]; __syncthreads(); }
  float var = red[0] * (1.0f / KD);
  float y = c * rsqrtf(var + 1e-5f) * gma[d] + bta[d];
  hf[(long)r * KD + d] = y;
  hb[(long)r * KD + d] = (bf16)y;
}

// ---------------- banded local attention (one block per (c, head, b)) ------
// Columns: col 0 = global score sg (q . k[S-1]); cols 1..150 = banded window.
__global__ __launch_bounds__(256)
void attn_local(const float* __restrict__ q, const float* __restrict__ k,
                const float* __restrict__ v, bf16* __restrict__ aout) {
  const int c = blockIdx.x, hh = blockIdx.y, b = blockIdx.z;
  const int tid = threadIdx.x;
  __shared__ float qc[WW * NHD];          // 12.8 KB
  __shared__ bf16  kc[152 * NHD];         // 19.5 KB (150 window rows + k[S-1])
  __shared__ float scr[WW * 152];         // 30.4 KB  (total 62.7 KB LDS)
  const long hd = (long)hh * NHD;

  for (int i = tid; i < WW * NHD; i += 256) {
    int qi = i >> 6, d = i & 63;
    qc[i] = q[((long)(b * SEQ + c * WW + qi)) * KD + hd + d];
  }
  for (int i = tid; i < 151 * NHD; i += 256) {
    int kj = i >> 6, d = i & 63;
    int jabs = (kj < 150) ? (c * WW - WW + kj) : (SEQ - 1);
    float kv = (jabs >= 0 && jabs < SEQ)
                   ? k[((long)(b * SEQ + jabs)) * KD + hd + d] : 0.0f;
    kc[kj * NHD + d] = (bf16)kv;
  }
  __syncthreads();

  for (int t = tid; t < WW * 151; t += 256) {
    int qi = t / 151, kj2 = t % 151;
    const float* qrow = qc + qi * NHD;
    float dot = 0.0f;
    if (kj2 < 150) {
      const bf16* krow = kc + kj2 * NHD;
      for (int d = 0; d < NHD; ++d) dot += qrow[d] * (float)krow[d];
      int jabs = c * WW - WW + kj2;
      int diff = kj2 - WW - qi; if (diff < 0) diff = -diff;
      bool band = (diff <= WW) && (jabs >= 0) && (jabs < SEQ);
      scr[qi * 152 + 1 + kj2] =
          band ? (dot + ((jabs == SEQ - 1) ? -10000.0f : 0.0f)) : -1.0e9f;
    } else {
      const bf16* krow = kc + 150 * NHD;
      for (int d = 0; d < NHD; ++d) dot += qrow[d] * (float)krow[d];
      scr[qi * 152 + 0] = dot;
    }
  }
  __syncthreads();

  if (tid < WW) {
    float mx = -3.0e38f;
    for (int j = 0; j < 151; ++j) mx = fmaxf(mx, scr[tid * 152 + j]);
    float sum = 0.0f;
    for (int j = 0; j < 151; ++j) {
      float e = expf(scr[tid * 152 + j] - mx);
      scr[tid * 152 + j] = e;
      sum += e;
    }
    float inv = 1.0f / sum;
    for (int j = 0; j < 151; ++j) scr[tid * 152 + j] *= inv;
  }
  __syncthreads();

  for (int t = tid; t < WW * NHD; t += 256) {
    int qi = t >> 6, d = t & 63;
    float acc = scr[qi * 152 + 0] * v[((long)(b * SEQ + SEQ - 1)) * KD + hd + d];
    for (int kj = 0; kj < 150; ++kj) {
      int jabs = c * WW - WW + kj;
      if (jabs >= 0 && jabs < SEQ)
        acc += scr[qi * 152 + 1 + kj] * v[((long)(b * SEQ + jabs)) * KD + hd + d];
    }
    aout[((long)(b * SEQ + c * WW + qi)) * KD + hd + d] = (bf16)acc;
  }
}

// ---------------- global-token attention (overwrites row S-1) --------------
__global__ __launch_bounds__(128)
void attn_global(const float* __restrict__ qg, const float* __restrict__ kg,
                 const float* __restrict__ vg, bf16* __restrict__ aout) {
  const int hh = blockIdx.x, b = blockIdx.y;
  const int tid = threadIdx.x;
  __shared__ float qs[NHD];
  __shared__ float p[SEQ];
  if (tid < NHD) qs[tid] = qg[(long)b * KD + hh * NHD + tid];
  __syncthreads();
  for (int s = tid; s < SEQ; s += 128) {
    const float* krow = kg + ((long)(b * SEQ + s)) * KD + hh * NHD;
    float dot = 0.0f;
    for (int d = 0; d < NHD; ++d) dot += qs[d] * krow[d];
    p[s] = dot;
  }
  __syncthreads();
  if (tid == 0) {
    float mx = -3.0e38f;
    for (int s = 0; s < SEQ; ++s) mx = fmaxf(mx, p[s]);
    float sum = 0.0f;
    for (int s = 0; s < SEQ; ++s) { float e = expf(p[s] - mx); p[s] = e; sum += e; }
    float inv = 1.0f / sum;
    for (int s = 0; s < SEQ; ++s) p[s] *= inv;
  }
  __syncthreads();
  if (tid < NHD) {
    float acc = 0.0f;
    for (int s = 0; s < SEQ; ++s)
      acc += p[s] * vg[((long)(b * SEQ + s)) * KD + hh * NHD + tid];
    aout[((long)(b * SEQ + SEQ - 1)) * KD + hh * NHD + tid] = (bf16)acc;
  }
}

// ---------------- final: xcls = LN(h[:,1099,:]); two linear heads ----------
__global__ __launch_bounds__(128)
void final_head(const float* __restrict__ hf, const float* __restrict__ fg,
                const float* __restrict__ fb, const float* __restrict__ we,
                const float* __restrict__ be, const float* __restrict__ wp,
                const float* __restrict__ bp, float* __restrict__ out) {
  int b = blockIdx.x, tid = threadIdx.x;
  __shared__ float x[KD];
  __shared__ float red[128];
  const float* row = hf + ((long)(b * SEQ + NC)) * KD;
  float ls = 0.0f;
  for (int d = tid; d < KD; d += 128) { float v = row[d]; x[d] = v; ls += v; }
  red[tid] = ls;
  __syncthreads();
  for (int o = 64; o > 0; o >>= 1) { if (tid < o) red[tid] += red[tid + o]; __syncthreads(); }
  float mean = red[0] * (1.0f / KD);
  __syncthreads();
  float ss = 0.0f;
  for (int d = tid; d < KD; d += 128) { float c = x[d] - mean; ss += c * c; }
  red[tid] = ss;
  __syncthreads();
  for (int o = 64; o > 0; o >>= 1) { if (tid < o) red[tid] += red[tid + o]; __syncthreads(); }
  float rstd = rsqrtf(red[0] * (1.0f / KD) + 1e-5f);
  __syncthreads();
  for (int d = tid; d < KD; d += 128) x[d] = (x[d] - mean) * rstd * fg[d] + fb[d];
  __syncthreads();
  if (tid < 11) {
    float acc = be[tid];
    for (int d = 0; d < KD; ++d) acc += x[d] * we[tid * KD + d];
    out[b * 11 + tid] = acc;
  } else if (tid >= 16 && tid < 21) {
    int o = tid - 16;
    float acc = bp[o];
    for (int d = 0; d < KD; ++d) acc += x[d] * wp[o * KD + d];
    out[88 + b * 5 + o] = acc;
  }
}

// ---------------------------------------------------------------------------
// Host launcher
// ---------------------------------------------------------------------------
static inline unsigned cdiv(unsigned a, unsigned b) { return (a + b - 1) / b; }

extern "C" void kernel_launch(void* const* d_in, const int* in_sizes, int n_in,
                              void* d_out, int out_size, void* d_ws, size_t ws_size,
                              hipStream_t stream) {
  (void)in_sizes; (void)n_in; (void)out_size; (void)ws_size;
  // Input order: ct, z, then params leaves in jax pytree (sorted-key) order.
  const float* ct = (const float*)d_in[0];
  // d_in[1] = z : unused by the reference forward pass.
  const float* b1  = (const float*)d_in[2];
  const float* b2  = (const float*)d_in[3];
  const float* be_ = (const float*)d_in[4];
  const float* bp_ = (const float*)d_in[5];
  const float* cls = (const float*)d_in[6];
  const float* eb  = (const float*)d_in[7];
  const float* eg  = (const float*)d_in[8];
  const float* fb  = (const float*)d_in[9];
  const float* fg  = (const float*)d_in[10];
  // layers: base 11 + 22*l ; sorted keys:
  // 0 bi,1 bk,2 bkg,3 bo,4 bo2,5 bq,6 bqg,7 bv,8 bvg,9 ln1b,10 ln1g,
  // 11 ln2b,12 ln2g,13 wi,14 wk,15 wkg,16 wo,17 wo2,18 wq,19 wqg,20 wv,21 wvg
  const float* pos = (const float*)d_in[55];
  const float* tok = (const float*)d_in[56];
  const float* w1  = (const float*)d_in[57];
  const float* w2  = (const float*)d_in[58];
  const float* we  = (const float*)d_in[59];
  const float* wp  = (const float*)d_in[60];

  const int M1 = NB * NC;   // 8792 summarizer rows
  const int M2 = NB * SEQ;  // 8800 transformer rows

  // ---- workspace bump allocator ----
  char* wsb = (char*)d_ws;
  size_t off = 0;
  auto alloc = [&](size_t bytes) -> void* {
    void* p = wsb + off;
    off = (off + bytes + 255) & ~(size_t)255;
    return p;
  };
  auto conv = [&](const float* src, bf16* dst, long n) {
    f2bf<<<dim3(cdiv((unsigned)n, 256)), dim3(256), 0, stream>>>(src, dst, n);
  };

  // bf16 weights
  bf16* w1b = (bf16*)alloc((size_t)256 * 128 * 2);
  bf16* w2b = (bf16*)alloc((size_t)512 * 256 * 2);
  bf16 *wqb[2], *wkb[2], *wvb[2], *wqgb[2], *wkgb[2], *wvgb[2], *wob[2], *wib[2], *wo2b[2];
  for (int l = 0; l < 2; ++l) {
    wqb[l]  = (bf16*)alloc((size_t)512 * 512 * 2);
    wkb[l]  = (bf16*)alloc((size_t)512 * 512 * 2);
    wvb[l]  = (bf16*)alloc((size_t)512 * 512 * 2);
    wqgb[l] = (bf16*)alloc((size_t)512 * 512 * 2);
    wkgb[l] = (bf16*)alloc((size_t)512 * 512 * 2);
    wvgb[l] = (bf16*)alloc((size_t)512 * 512 * 2);
    wob[l]  = (bf16*)alloc((size_t)512 * 512 * 2);
    wib[l]  = (bf16*)alloc((size_t)2048 * 512 * 2);
    wo2b[l] = (bf16*)alloc((size_t)512 * 2048 * 2);
  }
  // activations
  bf16*  cs_bf = (bf16*)alloc((size_t)M1 * 128 * 2);
  bf16*  s_bf  = (bf16*)alloc((size_t)M1 * 256 * 2);
  float* seq   = (float*)alloc((size_t)M1 * KD * 4);
  float* hf    = (float*)alloc((size_t)M2 * KD * 4);
  bf16*  hb    = (bf16*)alloc((size_t)M2 * KD * 2);
  float* q_f   = (float*)alloc((size_t)M2 * KD * 4);
  float* k_f   = (float*)alloc((size_t)M2 * KD * 4);
  float* v_f   = (float*)alloc((size_t)M2 * KD * 4);
  float* kg_f  = (float*)alloc((size_t)M2 * KD * 4);
  float* vg_f  = (float*)alloc((size_t)M2 * KD * 4);
  float* qg_f  = (float*)alloc((size_t)NB * KD * 4);
  bf16*  at_b  = (bf16*)alloc((size_t)M2 * KD * 2);
  float* a_f   = (float*)alloc((size_t)M2 * KD * 4);
  bf16*  m1_b  = (bf16*)alloc((size_t)M2 * 2048 * 2);
  float* m2_f  = (float*)alloc((size_t)M2 * KD * 4);

  // ---- weight conversion ----
  conv(w1, w1b, 256L * 128);
  conv(w2, w2b, 512L * 256);
  for (int l = 0; l < 2; ++l) {
    int base = 11 + 22 * l;
    conv((const float*)d_in[base + 18], wqb[l],  512L * 512);
    conv((const float*)d_in[base + 14], wkb[l],  512L * 512);
    conv((const float*)d_in[base + 20], wvb[l],  512L * 512);
    conv((const float*)d_in[base + 19], wqgb[l], 512L * 512);
    conv((const float*)d_in[base + 15], wkgb[l], 512L * 512);
    conv((const float*)d_in[base + 21], wvgb[l], 512L * 512);
    conv((const float*)d_in[base + 16], wob[l],  512L * 512);
    conv((const float*)d_in[base + 13], wib[l],  2048L * 512);
    conv((const float*)d_in[base + 17], wo2b[l], 512L * 2048);
  }

  // ---- 1) chunk means (TDM double-buffered) ----
  chunk_mean<<<dim3(M1), dim3(128), 0, stream>>>(ct, cs_bf, NC);

  // ---- 2,3) summarizer MLP (WMMA) ----
  gemm_bf16_wmma<<<dim3(cdiv(M1, 128), 256 / 64), dim3(256), 0, stream>>>(
      cs_bf, 128, w1b, 128, b1, nullptr, s_bf, 256, M1, 256, 128, 1.0f, ACT_RELU);
  gemm_bf16_wmma<<<dim3(cdiv(M1, 128), 512 / 64), dim3(256), 0, stream>>>(
      s_bf, 256, w2b, 256, b2, seq, nullptr, 512, M1, 512, 256, 1.0f, ACT_NONE);

  // ---- 4) embed + LN ----
  embed_ln<<<dim3(M2), dim3(KD), 0, stream>>>(seq, cls, pos, tok, eg, eb, hf, hb);

  // ---- 5) transformer layers ----
  const float qscale = 0.125f;  // 1/sqrt(64)
  for (int l = 0; l < 2; ++l) {
    int base = 11 + 22 * l;
    const float* bq  = (const float*)d_in[base + 5];
    const float* bk  = (const float*)d_in[base + 1];
    const float* bv  = (const float*)d_in[base + 7];
    const float* bqg = (const float*)d_in[base + 6];
    const float* bkg = (const float*)d_in[base + 2];
    const float* bvg = (const float*)d_in[base + 8];
    const float* bo  = (const float*)d_in[base + 3];
    const float* bi  = (const float*)d_in[base + 0];
    const float* bo2 = (const float*)d_in[base + 4];
    const float* l1g = (const float*)d_in[base + 10];
    const float* l1b = (const float*)d_in[base + 9];
    const float* l2g = (const float*)d_in[base + 12];
    const float* l2b = (const float*)d_in[base + 11];

    dim3 g512(cdiv(M2, 128), 512 / 64);
    gemm_bf16_wmma<<<g512, dim3(256), 0, stream>>>(hb, KD, wqb[l], KD, bq,
        q_f, nullptr, KD, M2, KD, KD, qscale, ACT_NONE);
    gemm_bf16_wmma<<<g512, dim3(256), 0, stream>>>(hb, KD, wkb[l], KD, bk,
        k_f, nullptr, KD, M2, KD, KD, 1.0f, ACT_NONE);
    gemm_bf16_wmma<<<g512, dim3(256), 0, stream>>>(hb, KD, wvb[l], KD, bv,
        v_f, nullptr, KD, M2, KD, KD, 1.0f, ACT_NONE);
    gemm_bf16_wmma<<<g512, dim3(256), 0, stream>>>(hb, KD, wkgb[l], KD, bkg,
        kg_f, nullptr, KD, M2, KD, KD, 1.0f, ACT_NONE);
    gemm_bf16_wmma<<<g512, dim3(256), 0, stream>>>(hb, KD, wvgb[l], KD, bvg,
        vg_f, nullptr, KD, M2, KD, KD, 1.0f, ACT_NONE);
    // qg: only the cls row per batch (A rows strided by 1100*512)
    gemm_bf16_wmma<<<dim3(1, 512 / 64), dim3(256), 0, stream>>>(
        hb + (long)NC * KD, SEQ * KD, wqgb[l], KD, bqg,
        qg_f, nullptr, KD, NB, KD, KD, qscale, ACT_NONE);

    attn_local<<<dim3(NCH, 8, NB), dim3(256), 0, stream>>>(q_f, k_f, v_f, at_b);
    attn_global<<<dim3(8, NB), dim3(128), 0, stream>>>(qg_f, kg_f, vg_f, at_b);

    gemm_bf16_wmma<<<g512, dim3(256), 0, stream>>>(at_b, KD, wob[l], KD, bo,
        a_f, nullptr, KD, M2, KD, KD, 1.0f, ACT_NONE);
    add_ln<<<dim3(M2), dim3(KD), 0, stream>>>(hf, a_f, l1g, l1b, hb);

    gemm_bf16_wmma<<<dim3(cdiv(M2, 128), 2048 / 64), dim3(256), 0, stream>>>(
        hb, KD, wib[l], KD, bi, nullptr, m1_b, 2048, M2, 2048, KD, 1.0f, ACT_GELU);
    gemm_bf16_wmma<<<g512, dim3(256), 0, stream>>>(m1_b, 2048, wo2b[l], 2048, bo2,
        m2_f, nullptr, KD, M2, KD, 2048, 1.0f, ACT_NONE);
    add_ln<<<dim3(M2), dim3(KD), 0, stream>>>(hf, m2_f, l2g, l2b, hb);
  }

  // ---- 6) final heads ----
  final_head<<<dim3(NB), dim3(128), 0, stream>>>(hf, fg, fb, we, be_, wp, bp_,
                                                 (float*)d_out);
}
